// Net_2095944040841
// MI455X (gfx1250) — matrix operand order repaired and checked by reading, not compiled, not run on hardware
//
#include <hip/hip_runtime.h>
#include <stdint.h>

// ---------------------------------------------------------------------------
// 3-layer LSTM (HID=50) over B=512, T=1024, IN=20, OUT=8 on gfx1250.
// Block = 128 threads (4 waves) owns a 16-row batch tile. Gate N-tiles are
// padded 13 -> 16 so each wave does exactly 4 tiles (branchless, same
// critical path). Gate GEMMs via v_wmma_f32_16x16x32_f16; weights live in
// LDS pre-swizzled fragment-major so each B fragment is 2x ds_load_b128;
// all of a tile's fragments are loaded before its WMMA chain.
// ---------------------------------------------------------------------------

#define HID     50
#define IN_DIM  20
#define OUT_DIM 8
#define GATES   200      // 4*HID
#define BATCH   512
#define TT      1024
#define NTP     16       // N-tiles padded (13 real) -> 4 per wave, branchless
#define GPITCH  256      // gbuf/bias column pitch (NTP*16)
#define ROWP    64       // padded row pitch for f32 state arrays
#define NWAVES  4
#define NTHR    (NWAVES * 32)
// per-layer fragment-major weight block: [nt][kc][lane][8 dwords]
#define LFRAG   (NTP * 4 * 32 * 8)    // 16384 dwords

typedef __attribute__((ext_vector_type(16))) _Float16 v16h;
typedef __attribute__((ext_vector_type(8)))  float    v8f;

__device__ __forceinline__ float sigf(float x) {
    return 1.0f / (1.0f + __expf(-x));
}
__device__ __forceinline__ float tanh_fast(float x) {
    return 2.0f / (1.0f + __expf(-2.0f * x)) - 1.0f;
}

__device__ __forceinline__ v8f wmma16(v16h a, v16h b, v8f c) {
    // D = A(16x32 f16) * B(32x16 f16) + C(16x16 f32)
    return __builtin_amdgcn_wmma_f32_16x16x32_f16(
        /*neg_a=*/false, a, /*neg_b=*/false, b,
        /*c_mod=*/(short)0, c, /*reuse_a=*/false, /*reuse_b=*/false);
}

// Contiguous 32-byte fragment load from LDS (-> 2x ds_load_b128).
__device__ __forceinline__ v16h load_frag(const uint32_t* base, int fragIdx) {
    const uint32_t* p = base + (size_t)fragIdx * 8;
    union { uint4 q[2]; v16h v; } u;
    u.q[0] = ((const uint4*)p)[0];
    u.q[1] = ((const uint4*)p)[1];
    return u.v;
}

// A fragment (16x32, k-chunk kc) built from f32 state in LDS.
// lane<16 holds K pairs {0..3, 8..11}, lane>=16 holds {4..7, 12..15}.
__device__ __forceinline__ v16h build_a_frag(const float* s0, int w0,
                                             const float* s1, int m, int kc) {
    v16h a;
    const int phi = (threadIdx.x & 16) ? 4 : 0;
#pragma unroll
    for (int v = 0; v < 8; ++v) {
        const int pi = ((v < 4) ? v : v + 4) + phi;
        const int K  = kc * 32 + 2 * pi;
#pragma unroll
        for (int s = 0; s < 2; ++s) {
            const int k = K + s;
            float val;
            if (k < 64) {
                val = (k < w0) ? s0[m * ROWP + k] : 0.0f;
            } else {
                const int k2 = k - 64;
                val = (k2 < HID) ? s1[m * ROWP + k2] : 0.0f;
            }
            a[2 * v + s] = (_Float16)val;
        }
    }
    return a;
}

// One layer's gate GEMM: 16 padded N-tiles, exactly 4 per wave, branchless.
// biasR = 4 per-wave hoisted bias values. skip1 (compile-time): K-chunk 1 of
// the A matrix is structurally zero (layer 1, slot0 width 20) -> elide.
__device__ __forceinline__ void layer_gemm(const uint32_t* bf, const float* biasR,
                                           const float* s0, int w0,
                                           const float* s1, float* gbuf,
                                           bool skip1) {
    const int lane31 = threadIdx.x & 31;
    const int lane15 = threadIdx.x & 15;
    const int hi     = (threadIdx.x & 16) ? 1 : 0;
    const int wave   = threadIdx.x >> 5;

    const v16h a0 = build_a_frag(s0, w0, s1, lane15, 0);
    v16h a1;
    if (!skip1) a1 = build_a_frag(s0, w0, s1, lane15, 1);
    const v16h a2 = build_a_frag(s0, w0, s1, lane15, 2);
    const v16h a3 = build_a_frag(s0, w0, s1, lane15, 3);

#pragma unroll
    for (int i = 0; i < 4; ++i) {
        const int nt = wave + i * NWAVES;          // 0..15, always valid
        const int n0 = nt * 16;
        const int fb = (nt * 4) * 32 + lane31;
        // batch all fragment loads ahead of the WMMA chain
        const v16h b0 = load_frag(bf, fb + 0 * 32);
        v16h b1;
        if (!skip1) b1 = load_frag(bf, fb + 1 * 32);
        const v16h b2 = load_frag(bf, fb + 2 * 32);
        const v16h b3 = load_frag(bf, fb + 3 * 32);

        const float bv = biasR[i];
        v8f acc = {bv, bv, bv, bv, bv, bv, bv, bv};
        acc = wmma16(a0, b0, acc);
        if (!skip1) acc = wmma16(a1, b1, acc);
        acc = wmma16(a2, b2, acc);
        acc = wmma16(a3, b3, acc);
#pragma unroll
        for (int r = 0; r < 8; ++r)
            gbuf[(r + hi * 8) * GPITCH + n0 + lane15] = acc[r];
    }
}

// Gate nonlinearity + state update. cin==nullptr means c_in = 0 (layer-3 bug).
__device__ __forceinline__ void lstm_elem(const float* gbuf, const float* cin,
                                          float* cout, float* hout) {
    for (int e = threadIdx.x; e < 16 * HID; e += NTHR) {
        const int row = e / HID;
        const int j   = e - row * HID;
        const float* g = gbuf + row * GPITCH;
        const float i_ = sigf(g[j]);
        const float f_ = sigf(g[HID + j]);
        const float gg = tanh_fast(g[2 * HID + j]);
        const float o_ = sigf(g[3 * HID + j]);
        const float c0 = cin ? cin[row * ROWP + j] : 0.0f;
        const float cn = f_ * c0 + i_ * gg;
        cout[row * ROWP + j] = cn;
        hout[row * ROWP + j] = o_ * tanh_fast(cn);
    }
}

extern "C" __global__ __launch_bounds__(NTHR)
void lstm3_persistent_wmma(const float* __restrict__ x,
                           const float* __restrict__ W1,  const float* __restrict__ b1,
                           const float* __restrict__ Wih1, const float* __restrict__ Whh1,
                           const float* __restrict__ bih1, const float* __restrict__ bhh1,
                           const float* __restrict__ Wih2, const float* __restrict__ Whh2,
                           const float* __restrict__ bih2, const float* __restrict__ bhh2,
                           const float* __restrict__ Wih3, const float* __restrict__ Whh3,
                           const float* __restrict__ bih3, const float* __restrict__ bhh3,
                           const float* __restrict__ W2,  const float* __restrict__ b2,
                           float* __restrict__ out) {
    extern __shared__ __align__(16) char smem[];
    // ---- LDS carve (dwords; every region 16B aligned) ----
    uint32_t* Bf    = (uint32_t*)smem;          // 3 * LFRAG  gate weights, fragment-major
    uint32_t* B2f   = Bf + 3 * LFRAG;           // 2*32*8     output-proj weights, frag-major
    float*    biasL = (float*)(B2f + 2 * 32 * 8); // 3*GPITCH
    float*    bias2 = biasL + 3 * GPITCH;       // 16
    float*    xBuf  = bias2 + 16;               // 16*ROWP
    float*    hA    = xBuf + 16 * ROWP;
    float*    hB    = hA + 16 * ROWP;
    float*    hC    = hB + 16 * ROWP;
    float*    cA    = hC + 16 * ROWP;
    float*    cB    = cA + 16 * ROWP;
    float*    gbuf  = cB + 16 * ROWP;           // 16*GPITCH

    const int tid    = threadIdx.x;
    const int lane15 = tid & 15;
    const int lane31 = tid & 31;
    const int hi     = (tid & 16) ? 1 : 0;
    const int wave   = tid >> 5;
    const int bbase  = blockIdx.x * 16;

    // ---------------- weight preload: fragment-major pair-packed f16 --------
    const float* WihL[3] = {Wih1, Wih2, Wih3};
    const float* WhhL[3] = {Whh1, Whh2, Whh3};
    const float* bihL[3] = {bih1, bih2, bih3};
    const float* bhhL[3] = {bhh1, bhh2, bhh3};

    for (int l = 0; l < 3; ++l) {
        uint32_t* bf = Bf + l * LFRAG;
        for (int e = tid; e < LFRAG; e += NTHR) {
            // layout [nt][kc][lane][v]
            const int v     = e & 7;
            const int lane_ = (e >> 3) & 31;
            const int kc    = (e >> 8) & 3;
            const int nt    = e >> 10;             // 0..15 (13..15 padded zero)
            const int n     = nt * 16 + (lane_ & 15);
            const int P     = kc * 16 + ((lane_ & 16) ? 4 : 0) + v + ((v < 4) ? 0 : 4);   // K-pair index
            float v0 = 0.0f, v1 = 0.0f;
            if (n < GATES) {
                if (P < 32) {                      // slot0: input projection
                    const int k = 2 * P;
                    if (l == 0) {                  // fused Wih1 @ W1  (K = IN_DIM)
                        if (k < IN_DIM) {
                            float a = 0.0f, bb = 0.0f;
                            for (int j = 0; j < HID; ++j) {
                                const float w = Wih1[n * HID + j];
                                a += w * W1[j * IN_DIM + k];
                                if (k + 1 < IN_DIM) bb += w * W1[j * IN_DIM + k + 1];
                            }
                            v0 = a; v1 = bb;
                        }
                    } else {                       // Wih.T  (K = HID)
                        if (k     < HID) v0 = WihL[l][n * HID + k];
                        if (k + 1 < HID) v1 = WihL[l][n * HID + k + 1];
                    }
                } else {                           // slot1 (K base 64): Whh.T
                    const int k = 2 * (P - 32);
                    if (k     < HID) v0 = WhhL[l][n * HID + k];
                    if (k + 1 < HID) v1 = WhhL[l][n * HID + k + 1];
                }
            }
            union { uint32_t u; _Float16 h[2]; } cv;
            cv.h[0] = (_Float16)v0; cv.h[1] = (_Float16)v1;
            bf[e] = cv.u;
        }
        for (int e = tid; e < GPITCH; e += NTHR) {
            float bv = 0.0f;
            if (e < GATES) {
                bv = bihL[l][e] + bhhL[l][e];
                if (l == 0) {                      // fold b1 through Wih1
                    float a = 0.0f;
                    for (int j = 0; j < HID; ++j) a += Wih1[e * HID + j] * b1[j];
                    bv += a;
                }
            }
            biasL[l * GPITCH + e] = bv;
        }
    }
    for (int e = tid; e < 2 * 32 * 8; e += NTHR) { // output projection, frag-major
        const int v     = e & 7;
        const int lane_ = (e >> 3) & 31;
        const int kc    = e >> 8;                  // 0..1 (K = 50 fits in chunks 0,1)
        const int n     = lane_ & 15;
        const int P     = kc * 16 + ((lane_ & 16) ? 4 : 0) + v + ((v < 4) ? 0 : 4);
        const int k     = 2 * P;
        float v0 = 0.0f, v1 = 0.0f;
        if (n < OUT_DIM) {
            if (k     < HID) v0 = W2[n * HID + k];
            if (k + 1 < HID) v1 = W2[n * HID + k + 1];
        }
        union { uint32_t u; _Float16 h[2]; } cv;
        cv.h[0] = (_Float16)v0; cv.h[1] = (_Float16)v1;
        B2f[e] = cv.u;
    }
    for (int e = tid; e < 16; e += NTHR) bias2[e] = (e < OUT_DIM) ? b2[e] : 0.0f;

    // zero-init recurrent state
    for (int e = tid; e < 16 * ROWP; e += NTHR)
        hA[e] = hB[e] = hC[e] = cA[e] = cB[e] = 0.0f;
    __syncthreads();

    // ---- hoist loop-invariants into registers ------------------------------
    float biasR[3][4];
#pragma unroll
    for (int l = 0; l < 3; ++l)
#pragma unroll
        for (int i = 0; i < 4; ++i) {
            const int nt = wave + i * NWAVES;
            biasR[l][i] = biasL[l * GPITCH + nt * 16 + lane15];
        }
    const v16h w2f0 = load_frag(B2f, 0 * 32 + lane31);
    const v16h w2f1 = load_frag(B2f, 1 * 32 + lane31);
    const float bv2 = bias2[lane15];

    // ---------------- time recurrence --------------------------------------
    for (int t = 0; t < TT; ++t) {
        // stage x_t tile [16 x IN_DIM]
        for (int e = tid; e < 16 * IN_DIM; e += NTHR) {
            const int row = e / IN_DIM, col = e - row * IN_DIM;
            xBuf[row * ROWP + col] =
                x[((size_t)(bbase + row) * TT + t) * IN_DIM + col];
        }
        if (t + 1 < TT && tid < 16)                // next-step prefetch
            __builtin_prefetch(&x[((size_t)(bbase + tid) * TT + (t + 1)) * IN_DIM], 0, 0);
        __syncthreads();

        // layer 1: gates = x @ (Wih1*W1).T + h1 @ Whh1.T + fused bias
        // (K-chunk 1 elided: slot0 width 20 -> A[:,32:64] == 0)
        layer_gemm(Bf, biasR[0], xBuf, IN_DIM, hA, gbuf, /*skip1=*/true);
        __syncthreads();
        lstm_elem(gbuf, cA, cA, hA);
        __syncthreads();

        // layer 2
        layer_gemm(Bf + LFRAG, biasR[1], hA, HID, hB, gbuf, /*skip1=*/false);
        __syncthreads();
        lstm_elem(gbuf, cB, cB, hB);
        __syncthreads();

        // layer 3 (reference bug preserved: reads c3==0, writes c2)
        layer_gemm(Bf + 2 * LFRAG, biasR[2], hB, HID, hC, gbuf, /*skip1=*/false);
        __syncthreads();
        lstm_elem(gbuf, nullptr, cB, hC);
        __syncthreads();

        // output projection: out_t = h3 @ W2.T + b2 (wave 0 only; EXEC all-1s)
        if (wave == 0) {
            const v16h a0 = build_a_frag(hC, HID, hC, lane15, 0);
            const v16h a1 = build_a_frag(hC, HID, hC, lane15, 1);
            v8f acc = {bv2, bv2, bv2, bv2, bv2, bv2, bv2, bv2};
            acc = wmma16(a0, w2f0, acc);
            acc = wmma16(a1, w2f1, acc);
            if (lane15 < OUT_DIM) {
#pragma unroll
                for (int r = 0; r < 8; ++r) {
                    const int row = r + hi * 8;
                    out[((size_t)(bbase + row) * TT + t) * OUT_DIM + lane15] = acc[r];
                }
            }
        }
        __syncthreads();
    }
}

extern "C" void kernel_launch(void* const* d_in, const int* in_sizes, int n_in,
                              void* d_out, int out_size, void* d_ws, size_t ws_size,
                              hipStream_t stream) {
    (void)in_sizes; (void)n_in; (void)out_size; (void)d_ws; (void)ws_size;
    const float* x    = (const float*)d_in[0];
    const float* W1   = (const float*)d_in[1];
    const float* b1   = (const float*)d_in[2];
    const float* Wih1 = (const float*)d_in[3];
    const float* Whh1 = (const float*)d_in[4];
    const float* bih1 = (const float*)d_in[5];
    const float* bhh1 = (const float*)d_in[6];
    const float* Wih2 = (const float*)d_in[7];
    const float* Whh2 = (const float*)d_in[8];
    const float* bih2 = (const float*)d_in[9];
    const float* bhh2 = (const float*)d_in[10];
    const float* Wih3 = (const float*)d_in[11];
    const float* Whh3 = (const float*)d_in[12];
    const float* bih3 = (const float*)d_in[13];
    const float* bhh3 = (const float*)d_in[14];
    const float* W2   = (const float*)d_in[15];
    const float* b2   = (const float*)d_in[16];

    // LDS dwords: 3*16384 weights + 512 out-proj + 3*256+16 bias
    //             + 6*1024 state/x + 16*256 gates = 60,688 dwords
    //             = 242,752 bytes (< 320 KB/WGP, 1 block per WGP)
    const size_t shmem = (size_t)(3 * LFRAG + 2 * 32 * 8) * 4 +
                         (size_t)(3 * GPITCH + 16) * 4 +
                         (size_t)(6 * 16 * ROWP) * 4 +
                         (size_t)(16 * GPITCH) * 4;

    dim3 grid(BATCH / 16);   // 32 independent recurrence chains (1 per WGP)
    dim3 block(NTHR);        // 4 waves fill the WGP's 4 SIMD32s
    lstm3_persistent_wmma<<<grid, block, shmem, stream>>>(
        x, W1, b1, Wih1, Whh1, bih1, bhh1, Wih2, Whh2, bih2, bhh2,
        Wih3, Whh3, bih3, bhh3, W2, b2, (float*)d_out);
}
